// CrossAttentionFormerBlock_1752346656897
// MI455X (gfx1250) — matrix operand order repaired and checked by reading, not compiled, not run on hardware
//
#include <hip/hip_runtime.h>

// ---------------------------------------------------------------------------
// CDNA5 (gfx1250) windowed cross-attention transformer block.
// f16 WMMA (v_wmma_f32_16x16x32_f16) for all GEMMs + flash attention,
// fp32 for LN / softmax / GELU. wave32 everywhere.
// Double-buffered LDS staging; async global->LDS copies when available.
// ---------------------------------------------------------------------------

typedef _Float16 half_t;
typedef __attribute__((ext_vector_type(16))) _Float16 v16h;
typedef __attribute__((ext_vector_type(8)))  float    v8f;
typedef __attribute__((ext_vector_type(2)))  _Float16 h2;
typedef __attribute__((ext_vector_type(8)))  _Float16 h8;   // 16-byte chunk

#define DIM   256
#define HEADS 8
#define HD    32          // head dim == WMMA K for f16 (16x16x32)
#define NTOK  32768
#define NWIN  64
#define NPW   512         // tokens per window (8^3)
#define POSN  3375        // 15^3 relative positions
#define MLPH  1024

// ---- optional CDNA5 async global->LDS copy (guarded; sync fallback) -------
#if defined(__has_builtin)
#if __has_builtin(__builtin_amdgcn_global_load_async_to_lds_b128) && \
    __has_builtin(__builtin_amdgcn_s_wait_asynccnt)
#define USE_ASYNC_COPY 1
#endif
#endif
#ifndef USE_ASYNC_COPY
#define USE_ASYNC_COPY 0
#endif

#if USE_ASYNC_COPY
// Builtin signature (per compiler diagnostic): first param is a pointer to
// int __attribute__((vector_size(16))) in the global (AS1 / __device__)
// address space; second is the same vector type in LDS (AS3).
typedef int v4i_ __attribute__((vector_size(16)));
typedef v4i_ __attribute__((address_space(1))) gv4i;
typedef v4i_ __attribute__((address_space(3))) lv4i;
static __device__ __forceinline__ void async_b128(const void* g, void* l) {
  // AS1 ptr: global VA is identity (inttoptr).  AS3 ptr: low 32 bits of the
  // flat LDS address are the wave-relative LDS byte offset (ISA 10.2).
  __builtin_amdgcn_global_load_async_to_lds_b128(
      (gv4i*)(unsigned long long)(size_t)g,
      (lv4i*)(size_t)(unsigned)(size_t)l, 0, 0);
}
static __device__ __forceinline__ void async_wait0() {
  __builtin_amdgcn_s_wait_asynccnt(0);
}
#endif

static __device__ __forceinline__ v8f wmma_f16(v16h a, v16h b, v8f c) {
  // D = A(16x32 f16) * B(32x16 f16) + C(16x16 f32)
  return __builtin_amdgcn_wmma_f32_16x16x32_f16(false, a, false, b, (short)0, c, false, false);
}

// token index -> window-partitioned row (win*512 + n)
static __device__ __forceinline__ int part_row(int l) {
  int h = l >> 10, w = (l >> 5) & 31, d = l & 31;
  int win = ((h >> 3) * 4 + (w >> 3)) * 4 + (d >> 3);
  int n = ((h & 7) << 6) + ((w & 7) << 3) + (d & 7);
  return win * NPW + n;
}

// A-fragment K index for VGPR v (contiguous order: k = 16*hi + 2*v, matches B fills)
static __device__ __forceinline__ int a_k0(int v, int hi) {
  return hi * 16 + 2 * v;
}

// ---------------------------------------------------------------------------
// f32 -> f16 cast (weights)
// ---------------------------------------------------------------------------
__global__ void cast_f32_f16_kernel(const float* __restrict__ src,
                                    half_t* __restrict__ dst, int n) {
  int i = blockIdx.x * 256 + threadIdx.x;
  if (i < n) dst[i] = (half_t)src[i];
}

// ---------------------------------------------------------------------------
// LayerNorm(x) and raw y, written in window-partitioned order as f16
// ---------------------------------------------------------------------------
__global__ __launch_bounds__(256) void ln_part_kernel(
    const float* __restrict__ x, const float* __restrict__ y,
    const float* __restrict__ g, const float* __restrict__ b,
    half_t* __restrict__ xw, half_t* __restrict__ yw) {
  int l = blockIdx.x, c = threadIdx.x;
  float v = x[(size_t)l * DIM + c];
  __shared__ float red[256];
  red[c] = v; __syncthreads();
  for (int s = 128; s > 0; s >>= 1) { if (c < s) red[c] += red[c + s]; __syncthreads(); }
  float mean = red[0] * (1.0f / DIM);
  __syncthreads();
  float d = v - mean;
  red[c] = d * d; __syncthreads();
  for (int s = 128; s > 0; s >>= 1) { if (c < s) red[c] += red[c + s]; __syncthreads(); }
  float rstd = rsqrtf(red[0] * (1.0f / DIM) + 1e-5f);
  int r = part_row(l);
  xw[(size_t)r * DIM + c] = (half_t)(d * rstd * g[c] + b[c]);
  yw[(size_t)r * DIM + c] = (half_t)y[(size_t)l * DIM + c];
}

// ---------------------------------------------------------------------------
// Generic WMMA GEMM: C_f16[M,N] = act((A_f16[M,K] @ B_f16[K,N] + bias) * scale)
// 128x128 block tile, 8 waves of 64x32, K-step 32, double-buffered LDS.
// M=32768 via grid.y=256.
// ---------------------------------------------------------------------------
__global__ __launch_bounds__(256) void gemm_wmma_f16(
    const half_t* __restrict__ A, const half_t* __restrict__ B,
    const float* __restrict__ bias, half_t* __restrict__ C,
    int K, int ldb, int ldc, float scale, int act) {
  __shared__ alignas(16) half_t As[2][128][40];   // [buf][m][k]
  __shared__ alignas(16) half_t Bt[2][128][40];   // [buf][n][k] transposed
  const int tid = threadIdx.x, lane = tid & 31, wave = tid >> 5;
  const int bm = blockIdx.y * 128, bn = blockIdx.x * 128;
  const int wm = (wave >> 2) * 64, wn = (wave & 3) * 32;
  const int l15 = lane & 15, hi = lane >> 4;

  // per-thread staging slots: A = 2 b128 chunks, B = 2 b128 chunks (scattered)
  const int a_row0 = tid >> 2;              // 0..63 (+64 for second chunk)
  const int a_cc   = (tid & 3) * 8;         // half offset within 32-wide k slab
  const int b_kr0  = tid >> 4;              // 0..15 (+16 for second chunk)
  const int b_cc   = (tid & 15) * 8;        // col offset (8 cols)
  const int nsteps = K >> 5;
  v8f acc[4][2] = {};

  // ---- prologue: stage k-step 0 into buffer 0
  {
#if USE_ASYNC_COPY
    async_b128(A + (size_t)(bm + a_row0) * K + a_cc, &As[0][a_row0][a_cc]);
    async_b128(A + (size_t)(bm + a_row0 + 64) * K + a_cc, &As[0][a_row0 + 64][a_cc]);
#else
    h8 va0 = *(const h8*)(A + (size_t)(bm + a_row0) * K + a_cc);
    h8 va1 = *(const h8*)(A + (size_t)(bm + a_row0 + 64) * K + a_cc);
#endif
    h8 vb0 = *(const h8*)(B + (size_t)(b_kr0) * ldb + bn + b_cc);
    h8 vb1 = *(const h8*)(B + (size_t)(b_kr0 + 16) * ldb + bn + b_cc);
#if !USE_ASYNC_COPY
    *(h8*)&As[0][a_row0][a_cc] = va0;
    *(h8*)&As[0][a_row0 + 64][a_cc] = va1;
#endif
#pragma unroll
    for (int j = 0; j < 8; ++j) Bt[0][b_cc + j][b_kr0] = vb0[j];
#pragma unroll
    for (int j = 0; j < 8; ++j) Bt[0][b_cc + j][b_kr0 + 16] = vb1[j];
#if USE_ASYNC_COPY
    async_wait0();
#endif
  }
  __syncthreads();

  for (int s = 0; s < nsteps; ++s) {
    const int buf = s & 1, nbuf = buf ^ 1;
    const int kkn = (s + 1) << 5;
    const bool more = (s + 1) < nsteps;

    // fragment loads from current buffer (merge into ds_load_b128 pairs)
    v16h a[4], b[2];
#pragma unroll
    for (int mt = 0; mt < 4; ++mt) {
      int row = wm + mt * 16 + l15;
#pragma unroll
      for (int v = 0; v < 8; ++v) {
        h2 t = *(const h2*)&As[buf][row][a_k0(v, hi)];
        a[mt][2 * v] = t[0]; a[mt][2 * v + 1] = t[1];
      }
    }
#pragma unroll
    for (int nt = 0; nt < 2; ++nt) {
      int col = wn + nt * 16 + l15;
#pragma unroll
      for (int v = 0; v < 8; ++v) {
        h2 t = *(const h2*)&Bt[buf][col][hi * 16 + 2 * v];
        b[nt][2 * v] = t[0]; b[nt][2 * v + 1] = t[1];
      }
    }

    // issue next-tile staging (overlaps the WMMA block below)
    h8 va0, va1, vb0, vb1;
    if (more) {
#if USE_ASYNC_COPY
      async_b128(A + (size_t)(bm + a_row0) * K + kkn + a_cc, &As[nbuf][a_row0][a_cc]);
      async_b128(A + (size_t)(bm + a_row0 + 64) * K + kkn + a_cc, &As[nbuf][a_row0 + 64][a_cc]);
#else
      va0 = *(const h8*)(A + (size_t)(bm + a_row0) * K + kkn + a_cc);
      va1 = *(const h8*)(A + (size_t)(bm + a_row0 + 64) * K + kkn + a_cc);
#endif
      vb0 = *(const h8*)(B + (size_t)(kkn + b_kr0) * ldb + bn + b_cc);
      vb1 = *(const h8*)(B + (size_t)(kkn + b_kr0 + 16) * ldb + bn + b_cc);
    }

#pragma unroll
    for (int mt = 0; mt < 4; ++mt)
#pragma unroll
      for (int nt = 0; nt < 2; ++nt)
        acc[mt][nt] = wmma_f16(a[mt], b[nt], acc[mt][nt]);

    if (more) {
#if !USE_ASYNC_COPY
      *(h8*)&As[nbuf][a_row0][a_cc] = va0;
      *(h8*)&As[nbuf][a_row0 + 64][a_cc] = va1;
#endif
#pragma unroll
      for (int j = 0; j < 8; ++j) Bt[nbuf][b_cc + j][b_kr0] = vb0[j];
#pragma unroll
      for (int j = 0; j < 8; ++j) Bt[nbuf][b_cc + j][b_kr0 + 16] = vb1[j];
#if USE_ASYNC_COPY
      async_wait0();
#endif
    }
    __syncthreads();
  }

  // epilogue: bias / scale / activation, f16 store
#pragma unroll
  for (int mt = 0; mt < 4; ++mt) {
#pragma unroll
    for (int nt = 0; nt < 2; ++nt) {
      int col = bn + wn + nt * 16 + l15;
      float bv = bias ? bias[col] : 0.0f;
#pragma unroll
      for (int r = 0; r < 8; ++r) {
        int row = bm + wm + mt * 16 + r + hi * 8;
        float v = (acc[mt][nt][r] + bv) * scale;
        if (act == 1) v = 0.5f * v * (1.0f + erff(v * 0.70710678118f)); // exact GELU
        C[(size_t)row * ldc + col] = (half_t)v;
      }
    }
  }
}

// ---------------------------------------------------------------------------
// Relative-position bias MLP: 3375 positions, 16-dim hidden, 8 heads out
// ---------------------------------------------------------------------------
static __device__ void ln16_relu(const float* a, float* t,
                                 const float* g, const float* b) {
  float m = 0.f;
#pragma unroll
  for (int j = 0; j < 16; ++j) m += a[j];
  m *= (1.0f / 16.0f);
  float v = 0.f;
#pragma unroll
  for (int j = 0; j < 16; ++j) { float d = a[j] - m; v += d * d; }
  float rs = rsqrtf(v * (1.0f / 16.0f) + 1e-5f);
#pragma unroll
  for (int j = 0; j < 16; ++j) {
    float u = (a[j] - m) * rs * g[j] + b[j];
    t[j] = u > 0.f ? u : 0.f;
  }
}

__global__ void pos_mlp_kernel(
    const float* pp_w, const float* pp_b,
    const float* g1, const float* b1, const float* w1, const float* bb1,
    const float* g2, const float* b2, const float* w2, const float* bb2,
    const float* g3, const float* b3, const float* w3, const float* bb3,
    float* __restrict__ posbuf) {
  int i = blockIdx.x * 256 + threadIdx.x;
  if (i >= POSN) return;
  float c0 = (float)(i / 225) - 7.f;
  float c1 = (float)((i / 15) % 15) - 7.f;
  float c2 = (float)(i % 15) - 7.f;
  float a[16], t[16];
#pragma unroll
  for (int j = 0; j < 16; ++j)
    a[j] = c0 * pp_w[j] + c1 * pp_w[16 + j] + c2 * pp_w[32 + j] + pp_b[j];
  ln16_relu(a, t, g1, b1);
#pragma unroll
  for (int j = 0; j < 16; ++j) {
    float s = bb1[j];
    for (int k = 0; k < 16; ++k) s += t[k] * w1[k * 16 + j];
    a[j] = s;
  }
  ln16_relu(a, t, g2, b2);
#pragma unroll
  for (int j = 0; j < 16; ++j) {
    float s = bb2[j];
    for (int k = 0; k < 16; ++k) s += t[k] * w2[k * 16 + j];
    a[j] = s;
  }
  ln16_relu(a, t, g3, b3);
#pragma unroll
  for (int h = 0; h < 8; ++h) {
    float s = bb3[h];
    for (int k = 0; k < 16; ++k) s += t[k] * w3[k * 8 + h];
    posbuf[(size_t)i * 8 + h] = s;
  }
}

// Gather pos -> bias table [head][n][m]  (8.4 MB, L2 resident for 64 windows)
__global__ void bias_gather_kernel(const float* __restrict__ posbuf,
                                   float* __restrict__ biastab) {
  int i = blockIdx.x * 256 + threadIdx.x;   // < 8*512*512
  int head = i >> 18;
  int n = (i >> 9) & 511, m = i & 511;
  int rh = (n >> 6) - (m >> 6) + 7;
  int rw = ((n >> 3) & 7) - ((m >> 3) & 7) + 7;
  int rd = (n & 7) - (m & 7) + 7;
  int pidx = (rh * 15 + rw) * 15 + rd;
  biastab[i] = posbuf[(size_t)pidx * 8 + head];
}

// ---------------------------------------------------------------------------
// Flash attention: block = (4 q-tiles x 8 heads x 64 windows), 8 waves,
// wave owns 16 q-rows; 32 keys per step; online softmax in registers.
// Double-buffered K/V chunks; half the threads stage K, half stage V^T.
// ---------------------------------------------------------------------------
__global__ __launch_bounds__(256) void attn_kernel(
    const half_t* __restrict__ qbuf,    // [32768,256] window order, pre-scaled
    const half_t* __restrict__ kvbuf,   // [32768,512] (k | v)
    const float*  __restrict__ biastab, // [8,512,512]
    half_t* __restrict__ att) {         // [32768,256]
  __shared__ alignas(16) half_t kc[2][32][40];   // K chunk  [key][hd]
  __shared__ alignas(16) half_t vtc[2][32][40];  // V chunk transposed [hd][key]
  __shared__ alignas(16) half_t pc[8][16][40];   // per-wave P bounce
  const int tid = threadIdx.x, lane = tid & 31, wave = tid >> 5;
  const int head = blockIdx.y, win = blockIdx.z;
  const int qbase = blockIdx.x * 128 + wave * 16;
  const int l15 = lane & 15, hi = lane >> 4;

  // staging: threads 0..127 copy K (b128 chunks), 128..255 copy+transpose V
  const int st_key = (tid & 127) >> 2;     // 0..31
  const int st_cc  = (tid & 3) * 8;        // 0,8,16,24 halves
  const bool is_k  = tid < 128;
  const half_t* kvrow0 =
      kvbuf + (size_t)(win * NPW + st_key) * (2 * DIM) + head * HD + st_cc;

  // Q A-fragment (hd = WMMA K = 32, loaded once)
  v16h aq;
  {
    const half_t* qp = qbuf + (size_t)(win * NPW + qbase + l15) * DIM + head * HD;
#pragma unroll
    for (int v = 0; v < 8; ++v) {
      h2 t = *(const h2*)(qp + a_k0(v, hi));
      aq[2 * v] = t[0]; aq[2 * v + 1] = t[1];
    }
  }

  v8f o0 = {}, o1 = {};
  float m[8], lsum[8];
#pragma unroll
  for (int r = 0; r < 8; ++r) { m[r] = -1e30f; lsum[r] = 0.f; }
  const float* bh = biastab + (size_t)head * NPW * NPW;

  // ---- prologue: stage kt=0 into buffer 0
  {
    if (is_k) {
#if USE_ASYNC_COPY
      async_b128(kvrow0, &kc[0][st_key][st_cc]);
#else
      h8 t = *(const h8*)kvrow0;
      *(h8*)&kc[0][st_key][st_cc] = t;
#endif
    } else {
      h8 t = *(const h8*)(kvrow0 + DIM);
#pragma unroll
      for (int j = 0; j < 8; ++j) vtc[0][st_cc + j][st_key] = t[j];
    }
#if USE_ASYNC_COPY
    async_wait0();
#endif
  }
  __syncthreads();

  for (int s = 0; s < 16; ++s) {           // 512 keys / 32
    const int kt = s * 32;
    const int buf = s & 1, nbuf = buf ^ 1;
    const bool more = s < 15;

    // S = q k^T for two 16-key tiles (B: hd pairs contiguous in kc rows)
    v16h bk0, bk1;
#pragma unroll
    for (int v = 0; v < 8; ++v) {
      int k0 = hi * 16 + 2 * v;
      h2 t0 = *(const h2*)&kc[buf][l15][k0];
      h2 t1 = *(const h2*)&kc[buf][16 + l15][k0];
      bk0[2 * v] = t0[0]; bk0[2 * v + 1] = t0[1];
      bk1[2 * v] = t1[0]; bk1[2 * v + 1] = t1[1];
    }

    // issue next-chunk staging (overlaps S/softmax/PV work below)
    h8 vv;
    if (more) {
      const half_t* kvn = kvrow0 + (size_t)(kt + 32) * (2 * DIM);
      if (is_k) {
#if USE_ASYNC_COPY
        async_b128(kvn, &kc[nbuf][st_key][st_cc]);
#else
        vv = *(const h8*)kvn;
#endif
      } else {
        vv = *(const h8*)(kvn + DIM);
      }
    }

    v8f zero = {};
    v8f s0 = wmma_f16(aq, bk0, zero);
    v8f s1 = wmma_f16(aq, bk1, zero);

    // bias + online softmax (row's 16 cols live in one 16-lane half)
    float p0[8], p1[8];
#pragma unroll
    for (int r = 0; r < 8; ++r) {
      int qr = qbase + r + hi * 8;
      float v0 = s0[r] + bh[(size_t)qr * NPW + kt + l15];
      float v1 = s1[r] + bh[(size_t)qr * NPW + kt + 16 + l15];
      float mx = fmaxf(v0, v1);
#pragma unroll
      for (int off = 1; off < 16; off <<= 1) mx = fmaxf(mx, __shfl_xor(mx, off, 32));
      float nm = fmaxf(m[r], mx);
      float al = __expf(m[r] - nm);
      float e0 = __expf(v0 - nm), e1 = __expf(v1 - nm);
      float rs = e0 + e1;
#pragma unroll
      for (int off = 1; off < 16; off <<= 1) rs += __shfl_xor(rs, off, 32);
      lsum[r] = lsum[r] * al + rs;
      m[r] = nm;
      o0[r] = o0[r] * al;
      o1[r] = o1[r] * al;
      p0[r] = e0; p1[r] = e1;
    }

    // P: C-layout -> A-layout via wave-private LDS bounce
#pragma unroll
    for (int r = 0; r < 8; ++r) {
      int row = r + hi * 8;
      pc[wave][row][l15] = (half_t)p0[r];
      pc[wave][row][16 + l15] = (half_t)p1[r];
    }
    asm volatile("s_wait_dscnt 0x0" ::: "memory");
    v16h ap;
#pragma unroll
    for (int v = 0; v < 8; ++v) {
      h2 t = *(const h2*)&pc[wave][l15][a_k0(v, hi)];
      ap[2 * v] = t[0]; ap[2 * v + 1] = t[1];
    }

    // O += P @ V   (B: key pairs contiguous in transposed vtc rows)
    v16h bv0, bv1;
#pragma unroll
    for (int v = 0; v < 8; ++v) {
      int k0 = hi * 16 + 2 * v;
      h2 t0 = *(const h2*)&vtc[buf][l15][k0];
      h2 t1 = *(const h2*)&vtc[buf][16 + l15][k0];
      bv0[2 * v] = t0[0]; bv0[2 * v + 1] = t0[1];
      bv1[2 * v] = t1[0]; bv1[2 * v + 1] = t1[1];
    }
    o0 = wmma_f16(ap, bv0, o0);
    o1 = wmma_f16(ap, bv1, o1);

    // complete next-chunk staging
    if (more) {
      if (is_k) {
#if !USE_ASYNC_COPY
        *(h8*)&kc[nbuf][st_key][st_cc] = vv;
#endif
      } else {
#pragma unroll
        for (int j = 0; j < 8; ++j) vtc[nbuf][st_cc + j][st_key] = vv[j];
      }
#if USE_ASYNC_COPY
      async_wait0();
#endif
    }
    __syncthreads();
  }

  // epilogue: normalize, store f16
#pragma unroll
  for (int r = 0; r < 8; ++r) {
    float inv = 1.0f / lsum[r];
    int qr = qbase + r + hi * 8;
    half_t* op = att + (size_t)(win * NPW + qr) * DIM + head * HD;
    op[l15] = (half_t)(o0[r] * inv);
    op[16 + l15] = (half_t)(o1[r] * inv);
  }
}

// ---------------------------------------------------------------------------
// Un-partition + residual + LN2 -> x2 (f16)
// ---------------------------------------------------------------------------
__global__ __launch_bounds__(256) void unpart_res_ln_kernel(
    const float* __restrict__ x, const half_t* __restrict__ projout,
    const float* __restrict__ g, const float* __restrict__ b,
    float* __restrict__ res1, half_t* __restrict__ x2) {
  int l = blockIdx.x, c = threadIdx.x;
  int r = part_row(l);
  float s = x[(size_t)l * DIM + c] + (float)projout[(size_t)r * DIM + c];
  res1[(size_t)l * DIM + c] = s;
  __shared__ float red[256];
  red[c] = s; __syncthreads();
  for (int st = 128; st > 0; st >>= 1) { if (c < st) red[c] += red[c + st]; __syncthreads(); }
  float mean = red[0] * (1.0f / DIM);
  __syncthreads();
  float d = s - mean;
  red[c] = d * d; __syncthreads();
  for (int st = 128; st > 0; st >>= 1) { if (c < st) red[c] += red[c + st]; __syncthreads(); }
  float rstd = rsqrtf(red[0] * (1.0f / DIM) + 1e-5f);
  x2[(size_t)l * DIM + c] = (half_t)(d * rstd * g[c] + b[c]);
}

__global__ void final_add_kernel(const float* __restrict__ res1,
                                 const half_t* __restrict__ f2,
                                 float* __restrict__ out) {
  size_t i = (size_t)blockIdx.x * 256 + threadIdx.x;
  out[i] = res1[i] + (float)f2[i];
}

// ---------------------------------------------------------------------------
// Host launch
// ---------------------------------------------------------------------------
extern "C" void kernel_launch(void* const* d_in, const int* in_sizes, int n_in,
                              void* d_out, int out_size, void* d_ws, size_t ws_size,
                              hipStream_t stream) {
  const float* x      = (const float*)d_in[0];
  const float* y      = (const float*)d_in[1];
  const float* n1_g   = (const float*)d_in[2];
  const float* n1_b   = (const float*)d_in[3];
  const float* qkv_w  = (const float*)d_in[4];
  const float* qkv_b  = (const float*)d_in[5];
  const float* pp_w   = (const float*)d_in[6];
  const float* pp_b   = (const float*)d_in[7];
  const float* p1_lng = (const float*)d_in[8];
  const float* p1_lnb = (const float*)d_in[9];
  const float* p1_w   = (const float*)d_in[10];
  const float* p1_b   = (const float*)d_in[11];
  const float* p2_lng = (const float*)d_in[12];
  const float* p2_lnb = (const float*)d_in[13];
  const float* p2_w   = (const float*)d_in[14];
  const float* p2_b   = (const float*)d_in[15];
  const float* p3_lng = (const float*)d_in[16];
  const float* p3_lnb = (const float*)d_in[17];
  const float* p3_w   = (const float*)d_in[18];
  const float* p3_b   = (const float*)d_in[19];
  const float* proj_w = (const float*)d_in[20];
  const float* proj_b = (const float*)d_in[21];
  const float* n2_g   = (const float*)d_in[22];
  const float* n2_b   = (const float*)d_in[23];
  const float* fc1_w  = (const float*)d_in[24];
  const float* fc1_b  = (const float*)d_in[25];
  const float* fc2_w  = (const float*)d_in[26];
  const float* fc2_b  = (const float*)d_in[27];
  // d_in[28..30] = H, W, D (fixed at 32)

  char* ws = (char*)d_ws;
  size_t off = 0;
  auto alloc = [&](size_t bytes) {
    size_t o = off; off += (bytes + 255) & ~(size_t)255; return o;
  };
  const size_t ACT16 = (size_t)NTOK * DIM * sizeof(half_t);        // 16 MB
  size_t o_xw  = alloc(ACT16);                                     // xw  -> att
  size_t o_yw  = alloc(ACT16);                                     // yw  -> projout
  size_t o_q   = alloc(ACT16);                                     // q   -> x2
  size_t o_kv  = alloc(2 * ACT16);                                 // kv  -> part of h
  size_t o_bt  = alloc((size_t)HEADS * NPW * NPW * 4);             // bias table
  size_t o_pos = alloc((size_t)POSN * HEADS * 4);
  size_t hneed = (size_t)NTOK * MLPH * sizeof(half_t);             // 64 MB
  if (off < o_kv + hneed) off = (o_kv + hneed + 255) & ~(size_t)255;
  size_t o_h    = o_kv;                                            // overlay dead kv/bias
  size_t o_res1 = alloc((size_t)NTOK * DIM * 4);
  size_t o_f2   = alloc(ACT16);
  size_t o_wqkv = alloc((size_t)DIM * 3 * DIM * sizeof(half_t));
  size_t o_wprj = alloc((size_t)DIM * DIM * sizeof(half_t));
  size_t o_wf1  = alloc((size_t)DIM * MLPH * sizeof(half_t));
  size_t o_wf2  = alloc((size_t)MLPH * DIM * sizeof(half_t));

  half_t* xw      = (half_t*)(ws + o_xw);
  half_t* att     = (half_t*)(ws + o_xw);
  half_t* yw      = (half_t*)(ws + o_yw);
  half_t* projout = (half_t*)(ws + o_yw);
  half_t* qb      = (half_t*)(ws + o_q);
  half_t* x2      = (half_t*)(ws + o_q);
  half_t* kv      = (half_t*)(ws + o_kv);
  half_t* hb      = (half_t*)(ws + o_h);
  float*  bt      = (float*)(ws + o_bt);
  float*  posb    = (float*)(ws + o_pos);
  float*  res1    = (float*)(ws + o_res1);
  half_t* f2      = (half_t*)(ws + o_f2);
  half_t* wqkv    = (half_t*)(ws + o_wqkv);
  half_t* wprj    = (half_t*)(ws + o_wprj);
  half_t* wf1     = (half_t*)(ws + o_wf1);
  half_t* wf2     = (half_t*)(ws + o_wf2);
  float*  out     = (float*)d_out;

  // weight casts
  cast_f32_f16_kernel<<<(DIM * 3 * DIM + 255) / 256, 256, 0, stream>>>(qkv_w, wqkv, DIM * 3 * DIM);
  cast_f32_f16_kernel<<<(DIM * DIM + 255) / 256, 256, 0, stream>>>(proj_w, wprj, DIM * DIM);
  cast_f32_f16_kernel<<<(DIM * MLPH + 255) / 256, 256, 0, stream>>>(fc1_w, wf1, DIM * MLPH);
  cast_f32_f16_kernel<<<(MLPH * DIM + 255) / 256, 256, 0, stream>>>(fc2_w, wf2, MLPH * DIM);

  // LN(x), y -> window order f16
  ln_part_kernel<<<NTOK, 256, 0, stream>>>(x, y, n1_g, n1_b, xw, yw);

  // q = (LN(x) @ Wq + bq) * hd^-0.5 ; kv = y @ W[:,256:768] + b[256:768]
  gemm_wmma_f16<<<dim3(2, 256), 256, 0, stream>>>(xw, wqkv, qkv_b, qb,
      DIM, 3 * DIM, DIM, 0.17677669529f, 0);
  gemm_wmma_f16<<<dim3(4, 256), 256, 0, stream>>>(yw, wqkv + DIM, qkv_b + DIM, kv,
      DIM, 3 * DIM, 2 * DIM, 1.0f, 0);

  // dynamic relative-position bias
  pos_mlp_kernel<<<(POSN + 255) / 256, 256, 0, stream>>>(
      pp_w, pp_b, p1_lng, p1_lnb, p1_w, p1_b,
      p2_lng, p2_lnb, p2_w, p2_b, p3_lng, p3_lnb, p3_w, p3_b, posb);
  bias_gather_kernel<<<(HEADS * NPW * NPW) / 256, 256, 0, stream>>>(posb, bt);

  // flash attention per (q-tile, head, window)
  attn_kernel<<<dim3(4, HEADS, NWIN), 256, 0, stream>>>(qb, kv, bt, att);

  // proj
  gemm_wmma_f16<<<dim3(2, 256), 256, 0, stream>>>(att, wprj, proj_b, projout,
      DIM, DIM, DIM, 1.0f, 0);

  // un-partition + residual + LN2
  unpart_res_ln_kernel<<<NTOK, 256, 0, stream>>>(x, projout, n2_g, n2_b, res1, x2);

  // MLP: fc1 (+GELU) then fc2
  gemm_wmma_f16<<<dim3(8, 256), 256, 0, stream>>>(x2, wf1, fc1_b, hb,
      DIM, MLPH, MLPH, 1.0f, 1);
  gemm_wmma_f16<<<dim3(2, 256), 256, 0, stream>>>(hb, wf2, fc2_b, f2,
      MLPH, DIM, DIM, 1.0f, 0);

  // final residual
  final_add_kernel<<<NTOK, 256, 0, stream>>>(res1, f2, out);
}
